// CrossSparseAggrNet_57449482551497
// MI455X (gfx1250) — matrix operand order repaired and checked by reading, not compiled, not run on hardware
//
#include <hip/hip_runtime.h>
#include <hip/hip_fp16.h>

// ---------------------------------------------------------------------------
// CrossSparseAggrNet on MI455X (gfx1250):
//   - all GEMMs via v_wmma_f32_16x16x32_f16
//   - X-tile staging in agg GEMM via Tensor Data Mover (tensor_load_to_lds)
// ---------------------------------------------------------------------------

typedef __attribute__((ext_vector_type(16))) _Float16 v16h;
typedef __attribute__((ext_vector_type(8)))  _Float16 v8h;
typedef __attribute__((ext_vector_type(8)))  float    v8f;
typedef unsigned int u32x4 __attribute__((ext_vector_type(4)));
typedef int          i32x4 __attribute__((ext_vector_type(4)));
typedef int          i32x8 __attribute__((ext_vector_type(8)));

#define NB    256   // batch
#define NTOK  196   // tokens per stream
#define DIMC  512   // channel dim
#define HIDC  102   // hidden
#define NCC   64    // num clusters
#define MW    40    // words
#define NPAD  208   // tokens padded to 16
#define HPAD  128   // hidden padded (K for GEMM2)
#define HT    112   // hidden padded to 16 (w1T rows)

#define HAS_TDM __has_builtin(__builtin_amdgcn_tensor_load_to_lds)

// ---- WMMA fragment loaders (CDNA5 16x16x32 f16 layouts, wave32) -----------
// A (16xK tile, row-major src [m][k]): lane m=lane&15, half=lane>>4
//   elems 0..7  -> k = half*8 .. half*8+7
//   elems 8..15 -> k = 16+half*8 .. 16+half*8+7
__device__ __forceinline__ v16h frag_A(const _Float16* base, int rowStride, int lane) {
  const int m = lane & 15, h = lane >> 4;
  union { v16h v; v8h p[2]; } u;
  const _Float16* q = base + (size_t)m * rowStride + h * 8;
  u.p[0] = *(const v8h*)(q);
  u.p[1] = *(const v8h*)(q + 16);
  return u.v;
}
// B (Kx16 tile, staged TRANSPOSED src [n][k]): lane n=lane&15, half=lane>>4
//   elems 0..7 -> k = half*8 .., elems 8..15 -> k = 16+half*8 ..
__device__ __forceinline__ v16h frag_Bt(const _Float16* base, int rowStride, int lane) {
  const int n = lane & 15, h = lane >> 4;
  union { v16h v; v8h p[2]; } u;
  const _Float16* q = base + (size_t)n * rowStride + h * 8;
  u.p[0] = *(const v8h*)(q);
  u.p[1] = *(const v8h*)(q + 16);
  return u.v;
}
__device__ __forceinline__ v8f wmma16(v16h a, v16h b, v8f c) {
  return __builtin_amdgcn_wmma_f32_16x16x32_f16(false, a, false, b, (short)0, c,
                                                false, false);
}
__device__ __forceinline__ v8f vzero8() {
  v8f a;
#pragma unroll
  for (int i = 0; i < 8; ++i) a[i] = 0.f;
  return a;
}
__device__ __forceinline__ float wave_sum(float v) {
#pragma unroll
  for (int m = 16; m >= 1; m >>= 1) v += __shfl_xor(v, m, 32);
  return v;
}
__device__ __forceinline__ float wave_max(float v) {
#pragma unroll
  for (int m = 16; m >= 1; m >>= 1) v = fmaxf(v, __shfl_xor(v, m, 32));
  return v;
}

#if HAS_TDM
// TDM: load a 2D f32 tile [tile_rows x 64] (row stride 512 elems) into LDS.
// Rows >= rows_avail are OOB -> zero-filled by the TDM (exact for padded
// tokens; their softmax weights are zero anyway).
// D# packing per CDNA5 ISA 8.3/8.4 (group0: count/lds/global/type,
// group1: data_size, tensor dims, tile dims, dim0 stride).
__device__ __forceinline__ void tdm_load_tile_f32(uint32_t ldsAddr,
                                                  const float* gptr,
                                                  uint32_t rows_avail) {
  const uint64_t ga = (uint64_t)(uintptr_t)gptr;
  const uint32_t td1 = rows_avail < 32u ? rows_avail : 32u;  // tensor_dim1
  u32x4 g0 = { 1u,                                        // count=1 (valid)
               ldsAddr,                                    // lds_addr bytes
               (uint32_t)ga,                               // global_addr lo32
               (uint32_t)((ga >> 32) & 0x01FFFFFFu) | (2u << 30) };  // hi + type=2
  i32x8 g1 = { (int)(2u << 16),          // wg_mask=0, data_size=4B (code 2)
               (int)(64u << 16),         // tensor_dim0 lo16 = 64
               (int)(td1 << 16),         // tensor_dim0 hi16=0 | tensor_dim1 lo16
               (int)(64u << 16),         // tensor_dim1 hi16=0 | tile_dim0 = 64
               (int)32,                  // tile_dim1 = 32, tile_dim2 = 0
               (int)512,                 // tensor_dim0_stride lo32 = 512
               0,                        // stride hi16 | dim1_stride lo16 = 0
               0 };                      // dim1_stride hi32 = 0
  i32x4 z4 = { 0, 0, 0, 0 };
#if __has_include(<hip/amd_detail/amd_gfx1250_TDM.h>)
  i32x8 z8 = { 0, 0, 0, 0, 0, 0, 0, 0 };
  __builtin_amdgcn_tensor_load_to_lds(g0, g1, z4, z4, z8, 0);   // clang-23 form
#else
  __builtin_amdgcn_tensor_load_to_lds(g0, g1, z4, z4, 0);       // ROCm 7.2 form
#endif
}
#endif

// ---------------------------------------------------------------------------
// Kernel 0: build f16 transposed/padded weights.
//   w1T[s][n(112)][k(512)]  (n>=102 -> 0)
//   w2T[s][n(64)][k(128)]   (k>=102 -> 0)
// ---------------------------------------------------------------------------
__global__ __launch_bounds__(256) void prep_weights(
    const float* __restrict__ w1s, const float* __restrict__ w2s,
    const float* __restrict__ w1d, const float* __restrict__ w2d,
    _Float16* __restrict__ w1T, _Float16* __restrict__ w2T) {
  const int W1N = 2 * HT * DIMC;
  const int W2N = 2 * NCC * HPAD;
  int idx = blockIdx.x * 256 + threadIdx.x;
  if (idx < W1N) {
    int s = idx / (HT * DIMC);
    int r = idx % (HT * DIMC);
    int n = r / DIMC, k = r % DIMC;
    const float* w1 = s ? w1d : w1s;
    w1T[idx] = (_Float16)((n < HIDC) ? w1[k * HIDC + n] : 0.f);
  } else if (idx < W1N + W2N) {
    int j = idx - W1N;
    int s = j / (NCC * HPAD);
    int r = j % (NCC * HPAD);
    int n = r / HPAD, k = r % HPAD;
    const float* w2 = s ? w2d : w2s;
    w2T[j] = (_Float16)((k < HIDC) ? w2[k * NCC + n] : 0.f);
  }
}

// ---------------------------------------------------------------------------
// Kernel 1: per 16-token tile: LayerNorm -> GEMM1(512->112,gelu) ->
//           GEMM2(112->64) -> logits f32 stored (s,b,nc,tok)
// grid = (13, B, 2), block = 256 (8 waves)
// ---------------------------------------------------------------------------
__global__ __launch_bounds__(256) void mlp_logits(
    const float* __restrict__ xs, const float* __restrict__ xd,
    const float* __restrict__ lngs, const float* __restrict__ lnbs,
    const float* __restrict__ b1s, const float* __restrict__ b2s,
    const float* __restrict__ lngd, const float* __restrict__ lnbd,
    const float* __restrict__ b1d, const float* __restrict__ b2d,
    const _Float16* __restrict__ w1T, const _Float16* __restrict__ w2T,
    const float* __restrict__ scale, float* __restrict__ logitsWS) {
  const int s = blockIdx.z;
  const int b = blockIdx.y;
  const int tok0 = blockIdx.x * 16;
  const float* x   = s ? xd : xs;
  const float* lng = s ? lngd : lngs;
  const float* lnb = s ? lnbd : lnbs;
  const float* b1  = s ? b1d : b1s;
  const float* b2  = s ? b2d : b2s;
  const _Float16* w1t = w1T + (size_t)s * (HT * DIMC);
  const _Float16* w2t = w2T + (size_t)s * (NCC * HPAD);
  float* logits = logitsWS + (size_t)s * ((size_t)NB * NCC * NPAD);

  __shared__ __align__(16) _Float16 lnx[16 * 520];   // LN'ed tokens, f16
  __shared__ __align__(16) _Float16 hbuf[16 * 136];  // hidden, f16, K-padded
  __shared__ float redS[16][17], redQ[16][17];
  __shared__ float muA[16], rsA[16];

  const int t  = threadIdx.x;
  const int tt = t >> 4;           // token within tile (16 thr / token)
  const int j  = t & 15;
  const int tok = tok0 + tt;
  const bool valid = (tok < NTOK);

  // ---- LayerNorm partials ----
  const float* xrow = x + ((size_t)b * NTOK + (valid ? tok : NTOK - 1)) * DIMC;
  float xl[32];
  float sum = 0.f, sq = 0.f;
#pragma unroll
  for (int i = 0; i < 8; ++i) {
    float4 v = *(const float4*)(xrow + j * 32 + i * 4);
    xl[i * 4 + 0] = v.x; xl[i * 4 + 1] = v.y;
    xl[i * 4 + 2] = v.z; xl[i * 4 + 3] = v.w;
    sum += v.x + v.y + v.z + v.w;
    sq  += v.x * v.x + v.y * v.y + v.z * v.z + v.w * v.w;
  }
  redS[tt][j] = sum; redQ[tt][j] = sq;
  // zero hidden buffer (covers padded K columns 112..135)
  for (int i = t; i < 16 * 136; i += 256) hbuf[i] = (_Float16)0.f;
  __syncthreads();
  if (j == 0) {
    float S = 0.f, Q = 0.f;
    for (int i = 0; i < 16; ++i) { S += redS[tt][i]; Q += redQ[tt][i]; }
    float mu = S * (1.f / DIMC);
    float var = Q * (1.f / DIMC) - mu * mu;
    muA[tt] = mu; rsA[tt] = rsqrtf(var + 1e-5f);
  }
  __syncthreads();
  {
    float mu = muA[tt], rs = rsA[tt];
#pragma unroll
    for (int i = 0; i < 32; ++i) {
      int c = j * 32 + i;
      float v = valid ? ((xl[i] - mu) * rs * lng[c] + lnb[c]) : 0.f;
      lnx[tt * 520 + c] = (_Float16)v;
    }
  }
  __syncthreads();

  const int w = t >> 5;
  const int lane = t & 31;
  const float scalev = scale[0];

  // ---- GEMM1: h = gelu(lnx @ w1 + b1), 7 hid tiles over waves 0..6 ----
  if (w < 7) {
    v8f acc = vzero8();
    for (int k0 = 0; k0 < DIMC; k0 += 32) {
      v16h a  = frag_A(lnx + k0, 520, lane);
      v16h bb = frag_Bt(w1t + (size_t)w * 16 * DIMC + k0, DIMC, lane);
      acc = wmma16(a, bb, acc);
    }
    const int hid = w * 16 + (lane & 15);
    const float b1v = (hid < HIDC) ? b1[hid] : 0.f;
    const int half = lane >> 4;
#pragma unroll
    for (int r = 0; r < 8; ++r) {
      float hv = acc[r] + b1v;
      hv = 0.5f * hv * (1.f + erff(hv * 0.70710678118654752f));  // exact gelu
      hbuf[(r + 8 * half) * 136 + hid] = (_Float16)hv;
    }
  }
  __syncthreads();

  // ---- GEMM2: logits = (h @ w2 + b2) * scale, 4 nc tiles over waves 0..3 ----
  if (w < 4) {
    v8f acc = vzero8();
    for (int k0 = 0; k0 < HPAD; k0 += 32) {
      v16h a  = frag_A(hbuf + k0, 136, lane);
      v16h bb = frag_Bt(w2t + (size_t)w * 16 * HPAD + k0, HPAD, lane);
      acc = wmma16(a, bb, acc);
    }
    const int nc = w * 16 + (lane & 15);
    const float b2v = b2[nc];
    const int half = lane >> 4;
    float* orow = logits + ((size_t)b * NCC + nc) * NPAD + tok0 + 8 * half;
#pragma unroll
    for (int r = 0; r < 8; ++r) orow[r] = (acc[r] + b2v) * scalev;
  }
}

// ---------------------------------------------------------------------------
// Kernel 2: softmax over tokens per (b,nc); writes f16 weights, zero-padded
// grid = (B, 2), block = 256 (8 waves x 8 rows)
// ---------------------------------------------------------------------------
__global__ __launch_bounds__(256) void softmax_rows(
    const float* __restrict__ logitsWS, _Float16* __restrict__ Wws) {
  const int b = blockIdx.x, s = blockIdx.y;
  const float* logits = logitsWS + (size_t)s * ((size_t)NB * NCC * NPAD);
  _Float16* Wout = Wws + (size_t)s * ((size_t)NB * NCC * NPAD);
  const int w = threadIdx.x >> 5, lane = threadIdx.x & 31;
  for (int rr = 0; rr < 8; ++rr) {
    const int nc = w * 8 + rr;
    const float* row = logits + ((size_t)b * NCC + nc) * NPAD;
    _Float16* orow = Wout + ((size_t)b * NCC + nc) * NPAD;
    float vals[7];
    float mx = -INFINITY;
#pragma unroll
    for (int i = 0; i < 7; ++i) {
      int idx = lane + i * 32;
      float v = (idx < NTOK) ? row[idx] : -INFINITY;
      vals[i] = v; mx = fmaxf(mx, v);
    }
    mx = wave_max(mx);
    float ssum = 0.f;
#pragma unroll
    for (int i = 0; i < 7; ++i) {
      int idx = lane + i * 32;
      float e = (idx < NTOK) ? __expf(vals[i] - mx) : 0.f;
      vals[i] = e; ssum += e;
    }
    ssum = wave_sum(ssum);
    const float inv = 1.f / ssum;
#pragma unroll
    for (int i = 0; i < 7; ++i) {
      int idx = lane + i * 32;
      if (idx < NPAD) orow[idx] = (_Float16)((idx < NTOK) ? vals[i] * inv : 0.f);
    }
  }
}

// ---------------------------------------------------------------------------
// Kernel 3: agg[b] = W_s[b] @ X_s[b] + W_d[b] @ X_d[b]  (64 x 512 per batch)
// grid = (8 c-tiles, B), block = 256 (8 waves)
// X chunk staged global->LDS by the Tensor Data Mover, then transposed to
// the f16 B-operand layout by the waves.
// ---------------------------------------------------------------------------
__global__ __launch_bounds__(256) void agg_gemm(
    const float* __restrict__ xs, const float* __restrict__ xd,
    const _Float16* __restrict__ Wws, _Float16* __restrict__ aggh) {
  const int b = blockIdx.y;
  const int c0 = blockIdx.x * 64;
  __shared__ __align__(16) float    Xraw[32 * 64];  // TDM destination (f32)
  __shared__ __align__(16) _Float16 Xt[64 * 40];    // [c][n] transposed f16
  const int t = threadIdx.x, w = t >> 5, lane = t & 31;
  const int mt = w & 3;           // nc tile
  const int cb = (w >> 2) * 2;    // first of 2 c-tiles for this wave
  v8f acc0 = vzero8(), acc1 = vzero8();

  for (int n0 = 0; n0 < NPAD; n0 += 32) {
    for (int s = 0; s < 2; ++s) {
      const float* x = s ? xd : xs;
      __syncthreads();   // previous chunk fully consumed
#if HAS_TDM
      if (w == 0) {
        tdm_load_tile_f32((uint32_t)(uintptr_t)Xraw,
                          x + ((size_t)b * NTOK + n0) * DIMC + c0,
                          (uint32_t)(NTOK - n0));
        __builtin_amdgcn_s_wait_tensorcnt(0);
      }
      __syncthreads();   // TDM data visible to all waves
      // transpose + f32->f16 convert: Xraw[n][c] -> Xt[c][n]
#pragma unroll
      for (int p = 0; p < 2; ++p) {
        int n = (t >> 4) + p * 16;
        int c4 = (t & 15) * 4;
        float4 v = *(const float4*)(&Xraw[n * 64 + c4]);
        Xt[(c4 + 0) * 40 + n] = (_Float16)v.x;
        Xt[(c4 + 1) * 40 + n] = (_Float16)v.y;
        Xt[(c4 + 2) * 40 + n] = (_Float16)v.z;
        Xt[(c4 + 3) * 40 + n] = (_Float16)v.w;
      }
#else
      // fallback: direct global staging with row clamp (weights are 0 there)
#pragma unroll
      for (int p = 0; p < 2; ++p) {
        int n = (t >> 4) + p * 16;
        int c4 = (t & 15) * 4;
        int nr = n0 + n; if (nr > NTOK - 1) nr = NTOK - 1;
        float4 v = *(const float4*)(x + ((size_t)b * NTOK + nr) * DIMC + c0 + c4);
        Xt[(c4 + 0) * 40 + n] = (_Float16)v.x;
        Xt[(c4 + 1) * 40 + n] = (_Float16)v.y;
        Xt[(c4 + 2) * 40 + n] = (_Float16)v.z;
        Xt[(c4 + 3) * 40 + n] = (_Float16)v.w;
      }
#endif
      __syncthreads();
      const _Float16* Wrow = Wws + (size_t)s * ((size_t)NB * NCC * NPAD) +
                             ((size_t)b * NCC + mt * 16) * NPAD + n0;
      if (n0 + 32 < NPAD) __builtin_prefetch(Wrow + 32, 0, 1);  // next W chunk
      v16h a   = frag_A(Wrow, NPAD, lane);
      v16h bb0 = frag_Bt(Xt + cb * 16 * 40, 40, lane);
      v16h bb1 = frag_Bt(Xt + (cb + 1) * 16 * 40, 40, lane);
      acc0 = wmma16(a, bb0, acc0);
      acc1 = wmma16(a, bb1, acc1);
    }
  }
  const int half = lane >> 4, n = lane & 15;
#pragma unroll
  for (int r = 0; r < 8; ++r) {
    int nc = mt * 16 + r + 8 * half;
    aggh[((size_t)b * NCC + nc) * DIMC + c0 + cb * 16 + n]       = (_Float16)acc0[r];
    aggh[((size_t)b * NCC + nc) * DIMC + c0 + (cb + 1) * 16 + n] = (_Float16)acc1[r];
  }
}

// ---------------------------------------------------------------------------
// Kernel 4: l2norm (folded as post-scale) + sim GEMM (40x512x64) +
//           leaky_relu + max/mean pooling -> out[b]
// grid = B, block = 256 (8 waves)
// ---------------------------------------------------------------------------
__global__ __launch_bounds__(256) void sim_pool(
    const float* __restrict__ words, const _Float16* __restrict__ aggh,
    float* __restrict__ out) {
  const int b = blockIdx.x;
  __shared__ __align__(16) _Float16 wh[48 * 520];  // words f16, padded rows
  __shared__ float simL[48 * 68];
  __shared__ float vn[64], tn[48];
  __shared__ float rm[MW], cm[64];
  const int t = threadIdx.x, w = t >> 5, lane = t & 31;

  // ---- stage words -> f16 LDS (rows >= 40 zeroed) ----
#pragma unroll
  for (int p = 0; p < 3; ++p) {
    int m = (t >> 4) + p * 16;
    int cj = (t & 15) * 32;
    const bool mv = (m < MW);
    const float* wr = words + ((size_t)b * MW + (mv ? m : 0)) * DIMC;
#pragma unroll
    for (int i = 0; i < 8; ++i) {
      float4 v = *(const float4*)(wr + cj + i * 4);
      wh[m * 520 + cj + i * 4 + 0] = (_Float16)(mv ? v.x : 0.f);
      wh[m * 520 + cj + i * 4 + 1] = (_Float16)(mv ? v.y : 0.f);
      wh[m * 520 + cj + i * 4 + 2] = (_Float16)(mv ? v.z : 0.f);
      wh[m * 520 + cj + i * 4 + 3] = (_Float16)(mv ? v.w : 0.f);
    }
  }
  // ---- v norms (64 rows from agg f16) ----
  for (int rr = 0; rr < 8; ++rr) {
    int nc = w * 8 + rr;
    const _Float16* ar = aggh + ((size_t)b * NCC + nc) * DIMC;
    float sqs = 0.f;
#pragma unroll
    for (int i = 0; i < 16; ++i) {
      float v = (float)ar[lane + i * 32];
      sqs += v * v;
    }
    sqs = wave_sum(sqs);
    if (lane == 0) vn[nc] = fmaxf(sqrtf(sqs), 1e-8f);
  }
  // ---- t norms (40 rows from words f32; pad rows get 1.0) ----
  for (int rr = 0; rr < 8; ++rr) {
    int m = w * 8 + rr;
    if (m < MW) {
      const float* wr = words + ((size_t)b * MW + m) * DIMC;
      float sqs = 0.f;
#pragma unroll
      for (int i = 0; i < 16; ++i) {
        float v = wr[lane + i * 32];
        sqs += v * v;
      }
      sqs = wave_sum(sqs);
      if (lane == 0) tn[m] = fmaxf(sqrtf(sqs), 1e-8f);
    } else if (m < 48 && lane == 0) {
      tn[m] = 1.f;
    }
  }
  __syncthreads();

  // ---- sim GEMM: 12 tiles (3 m-tiles x 4 nc-tiles) over 8 waves ----
  for (int tile = w; tile < 12; tile += 8) {
    const int mt = tile >> 2, nt = tile & 3;
    v8f acc = vzero8();
    for (int k0 = 0; k0 < DIMC; k0 += 32) {
      v16h a  = frag_A(wh + mt * 16 * 520 + k0, 520, lane);
      v16h bb = frag_Bt(aggh + ((size_t)b * NCC + nt * 16) * DIMC + k0, DIMC, lane);
      acc = wmma16(a, bb, acc);
    }
    const int half = lane >> 4, n = nt * 16 + (lane & 15);
#pragma unroll
    for (int r = 0; r < 8; ++r) {
      int m = mt * 16 + r + 8 * half;
      float v = acc[r] / (tn[m] * vn[n]);   // fold both l2norms
      v = (v >= 0.f) ? v : 0.1f * v;        // leaky_relu(0.1)
      simL[m * 68 + n] = v;
    }
  }
  __syncthreads();

  // ---- pooling: row max->mean (word side) + col max->mean (patch side) ----
  if (t < MW) {
    float mx = -INFINITY;
    for (int k = 0; k < 64; ++k) mx = fmaxf(mx, simL[t * 68 + k]);
    rm[t] = mx;
  }
  if (t >= 64 && t < 128) {
    int k = t - 64;
    float mx = -INFINITY;
    for (int m = 0; m < MW; ++m) mx = fmaxf(mx, simL[m * 68 + k]);
    cm[k] = mx;
  }
  __syncthreads();
  if (t == 0) {
    float a = 0.f, c = 0.f;
    for (int i = 0; i < MW; ++i) a += rm[i];
    for (int i = 0; i < 64; ++i) c += cm[i];
    out[b] = a * (1.f / MW) + c * (1.f / 64.f);
  }
}

// ---------------------------------------------------------------------------
extern "C" void kernel_launch(void* const* d_in, const int* in_sizes, int n_in,
                              void* d_out, int out_size, void* d_ws, size_t ws_size,
                              hipStream_t stream) {
  const float* xs    = (const float*)d_in[0];
  const float* xd    = (const float*)d_in[1];
  const float* words = (const float*)d_in[2];
  const float* lngs  = (const float*)d_in[3];
  const float* lnbs  = (const float*)d_in[4];
  const float* w1s   = (const float*)d_in[5];
  const float* b1s   = (const float*)d_in[6];
  const float* w2s   = (const float*)d_in[7];
  const float* b2s   = (const float*)d_in[8];
  const float* lngd  = (const float*)d_in[9];
  const float* lnbd  = (const float*)d_in[10];
  const float* w1d   = (const float*)d_in[11];
  const float* b1d   = (const float*)d_in[12];
  const float* w2d   = (const float*)d_in[13];
  const float* b2d   = (const float*)d_in[14];
  const float* scale = (const float*)d_in[15];
  float* out = (float*)d_out;

  // workspace layout (bytes, all 16B aligned)
  char* ws = (char*)d_ws;
  const size_t OFF_W1T  = 0;                               // 2*112*512 f16
  const size_t OFF_W2T  = OFF_W1T + (size_t)2 * HT * DIMC * 2;     // 229376
  const size_t OFF_LOG  = OFF_W2T + (size_t)2 * NCC * HPAD * 2;    // 262144
  const size_t OFF_WF16 = OFF_LOG + (size_t)2 * NB * NCC * NPAD * 4;
  const size_t OFF_AGG  = OFF_WF16 + (size_t)2 * NB * NCC * NPAD * 2;
  _Float16* w1T    = (_Float16*)(ws + OFF_W1T);
  _Float16* w2T    = (_Float16*)(ws + OFF_W2T);
  float*    logits = (float*)(ws + OFF_LOG);
  _Float16* Wf16   = (_Float16*)(ws + OFF_WF16);
  _Float16* aggh   = (_Float16*)(ws + OFF_AGG);

  prep_weights<<<dim3(512), dim3(256), 0, stream>>>(w1s, w2s, w1d, w2d, w1T, w2T);

  mlp_logits<<<dim3(13, NB, 2), dim3(256), 0, stream>>>(
      xs, xd, lngs, lnbs, b1s, b2s, lngd, lnbd, b1d, b2d, w1T, w2T, scale, logits);

  softmax_rows<<<dim3(NB, 2), dim3(256), 0, stream>>>(logits, Wf16);

  agg_gemm<<<dim3(8, NB), dim3(256), 0, stream>>>(xs, xd, Wf16, aggh);

  sim_pool<<<dim3(NB), dim3(256), 0, stream>>>(words, aggh, out);
}
